// AdaptiveTemporalAttentionModel_13915694039790
// MI455X (gfx1250) — matrix operand order repaired and checked by reading, not compiled, not run on hardware
//
#include <hip/hip_runtime.h>
#include <hip/hip_bf16.h>

typedef __attribute__((ext_vector_type(16))) _Float16 v16h;
typedef __attribute__((ext_vector_type(8)))  _Float16 v8h;
typedef __attribute__((ext_vector_type(8)))  float    v8f;
typedef unsigned int u32x4 __attribute__((ext_vector_type(4)));
typedef int          i32x8 __attribute__((ext_vector_type(8)));
typedef int          i32x4 __attribute__((ext_vector_type(4)));

// ---------------------------------------------------------------------------
// WMMA helper (CDNA5 / gfx1250, wave32). D = A(16x32 f16) * B(32x16 f16) + C
// ---------------------------------------------------------------------------
__device__ __forceinline__ v8f wmma16(v16h a, v16h b, v8f c) {
  return __builtin_amdgcn_wmma_f32_16x16x32_f16(
      false, a, false, b, (short)0, c, false, false);
}

// A fragment (16x32 f16) from f16 LDS: lane holds row m = lane&15; lanes 0-15
// carry K groups [0..7],[16..23]; lanes 16-31 carry [8..15],[24..31].
// Groups of 8 contiguous halfs -> compiler merges into ds_load_b128 pairs.
__device__ __forceinline__ v16h frag_a_lds(const _Float16* S, int ld,
                                           int row0, int k0, int lane) {
  int m = lane & 15, hi = lane >> 4;
  v16h a;
#pragma unroll
  for (int j = 0; j < 16; ++j) {
    int k = (j < 8) ? (8 * hi + j) : (16 + 8 * hi + (j - 8));
    a[j] = S[(row0 + m) * ld + k0 + k];
  }
  return a;
}

// Same fragment but sourced from an f32 LDS buffer (TDM-staged x), converting
// on the fly.
__device__ __forceinline__ v16h frag_a_lds_f32(const float* S, int ld,
                                               int row0, int k0, int lane) {
  int m = lane & 15, hi = lane >> 4;
  v16h a;
#pragma unroll
  for (int j = 0; j < 16; ++j) {
    int k = (j < 8) ? (8 * hi + j) : (16 + 8 * hi + (j - 8));
    a[j] = (_Float16)S[(row0 + m) * ld + k0 + k];
  }
  return a;
}

// A fragment gathered from zero-padded h0 (50 rows x 64): the K index of the
// 48x192 conv-as-GEMM selects the temporal shift (im2col without materializing).
__device__ __forceinline__ v16h frag_a_conv(const _Float16* h0p,
                                            int row0, int k0, int lane) {
  int m = lane & 15, hi = lane >> 4;
  v16h a;
#pragma unroll
  for (int j = 0; j < 16; ++j) {
    int kj = (j < 8) ? (8 * hi + j) : (16 + 8 * hi + (j - 8));
    int kk = k0 + kj;
    int shift = kk >> 6;
    int col   = kk & 63;
    a[j] = h0p[(row0 + m + shift) * 64 + col];
  }
  return a;
}

// B fragment from pre-swizzled weights: fragment (kc,nt) is stored
// lane-major/contiguous, so each lane issues two coalesced b128 loads.
__device__ __forceinline__ v16h frag_b_sw(const _Float16* Wsw, int NT,
                                          int kc, int nt, int lane) {
  const _Float16* p = Wsw + ((((kc * NT) + nt) * 32 + lane) << 4);
  v8h lo = *(const v8h*)(p);
  v8h hi = *(const v8h*)(p + 8);
  return __builtin_shufflevector(lo, hi, 0, 1, 2, 3, 4, 5, 6, 7,
                                 8, 9, 10, 11, 12, 13, 14, 15);
}

// ---------------------------------------------------------------------------
// Kernel 0: weight prep. f32 -> f16, rearranged into B-fragment order:
//   dst[ ((kc*NT + nt)*32 + lane)*16 + j ] = W[(kc*32 + (j<8 ? 8*(lane>>4)+j : 16+8*(lane>>4)+j-8))*N + nt*16 + (lane&15)]
// ---------------------------------------------------------------------------
__device__ __forceinline__ void swizzle_one(const float* src, _Float16* dst,
                                            int K, int N, int i, int stride) {
  int NT = N >> 4;
  int total = K * N;
  for (int o = i; o < total; o += stride) {
    int j = o & 15;
    int lane = (o >> 4) & 31;
    int f = o >> 9;
    int nt = f % NT;
    int kc = f / NT;
    int k = kc * 32 + ((j < 8) ? (8 * (lane >> 4) + j) : (16 + 8 * (lane >> 4) + (j - 8)));
    int n = nt * 16 + (lane & 15);
    dst[o] = (_Float16)src[k * N + n];
  }
}

__global__ void prep_weights(const float* W_in, const float* W_conv,
                             const float* Wq, const float* Wk, const float* Wv,
                             const float* W1, const float* W2, const float* W3,
                             _Float16* p_in, _Float16* p_conv, _Float16* p_q,
                             _Float16* p_k, _Float16* p_v, _Float16* p1,
                             _Float16* p2, _Float16* p3) {
  int i = blockIdx.x * blockDim.x + threadIdx.x;
  int stride = gridDim.x * blockDim.x;
  swizzle_one(W_in, p_in, 32, 64, i, stride);
  swizzle_one(W_conv, p_conv, 192, 64, i, stride);  // (KT,C0,C1) flat == 192x64
  swizzle_one(Wq, p_q, 64, 64, i, stride);
  swizzle_one(Wk, p_k, 64, 64, i, stride);
  swizzle_one(Wv, p_v, 64, 64, i, stride);
  swizzle_one(W1, p1, 64, 256, i, stride);
  swizzle_one(W2, p2, 256, 128, i, stride);
  // W3: 128x12 zero-padded to 128x16, then swizzled (NT == 1)
  for (int o = i; o < 128 * 16; o += stride) {
    int j = o & 15;
    int lane = (o >> 4) & 31;
    int kc = o >> 9;
    int k = kc * 32 + ((j < 8) ? (8 * (lane >> 4) + j) : (16 + 8 * (lane >> 4) + (j - 8)));
    int n = lane & 15;
    p3[o] = (_Float16)((n < 12) ? W3[k * 12 + n] : 0.f);
  }
}

// ---------------------------------------------------------------------------
// Kernel 1: fused input GEMM + conv + k/v (+ q last tile) + row-47 attention.
// One wave per sequence; 2 waves per block (static LDS ~62.8KB).
// x staged into LDS by the Tensor Data Mover (TDM), f32 raw.
// ---------------------------------------------------------------------------
#define WAVES_A 2

__global__ __launch_bounds__(32 * WAVES_A)
void attn_fused(const float* __restrict__ x,
                const float* __restrict__ b_in, const float* __restrict__ b_conv,
                const float* __restrict__ bq, const float* __restrict__ bk,
                const float* __restrict__ bv,
                const _Float16* __restrict__ Wp_in,
                const _Float16* __restrict__ Wp_conv,
                const _Float16* __restrict__ Wp_q,
                const _Float16* __restrict__ Wp_k,
                const _Float16* __restrict__ Wp_v,
                float* __restrict__ ctxbuf) {
  __shared__ float    xf32[WAVES_A][48 * 32];  // TDM destination (raw f32)
  __shared__ _Float16 h0p[WAVES_A][50 * 64];   // rows 0 and 49 are zero pad
  __shared__ _Float16 h1s[WAVES_A][48 * 64];
  __shared__ _Float16 kbs[WAVES_A][48 * 64];
  __shared__ _Float16 vbs[WAVES_A][48 * 64];
  __shared__ float    qls[WAVES_A][64];
  __shared__ float    ats[WAVES_A][48];

  const int lane = threadIdx.x & 31;
  const int wv   = threadIdx.x >> 5;
  const int hi   = lane >> 4;
  const int n15  = lane & 15;
  const size_t seq = (size_t)blockIdx.x * WAVES_A + wv;

  const float* xseq = x + seq * 48 * 32;
  float*    xw = xf32[wv];
  _Float16* h0 = h0p[wv];
  _Float16* h1 = h1s[wv];
  _Float16* kb = kbs[wv];
  _Float16* vb = vbs[wv];

  // Warm L2 for the streamed weight fragments.
  if (threadIdx.x < 12) __builtin_prefetch(Wp_conv + (int)threadIdx.x * 1024, 0, 1);

  // stage 0: x -> LDS via Tensor Data Mover (1-D tile, 1536 f32 elements)
#if defined(__has_builtin) && __has_builtin(__builtin_amdgcn_tensor_load_to_lds)
  {
    unsigned long long gaddr = (unsigned long long)(const void*)xseq;
    unsigned lds_addr = (unsigned)(unsigned long long)(void*)xw;
    u32x4 g0 = {
        0x1u,                                   // count=1, user mode, no gather
        lds_addr,                               // lds_addr [63:32]
        (unsigned)(gaddr & 0xffffffffu),        // global_addr [95:64]
        (unsigned)(((gaddr >> 32) & 0x1ffffffu) | 0x80000000u)  // addr hi | type=2
    };
    i32x8 g1 = {
        0x00020000,  // data_size = 2 (4 bytes)
        0x06000000,  // tensor_dim0 = 1536 (bits 79:48)
        0x00010000,  // tensor_dim1 = 1
        0x06000000,  // tile_dim0 = 1536 (bits 127:112)
        0x00000001,  // tile_dim1 = 1
        1536,        // tensor_dim0_stride (bits 207:160)
        0, 0
    };
    i32x4 gz4 = {0, 0, 0, 0};
    i32x8 gz8 = {0, 0, 0, 0, 0, 0, 0, 0};
    __builtin_amdgcn_tensor_load_to_lds(g0, g1, gz4, gz4, gz8, 0);
  }
#else
  for (int i = lane; i < 48 * 32; i += 32) xw[i] = xseq[i];
#endif

  // zero the conv pad rows meanwhile
  for (int i = lane; i < 64; i += 32) {
    h0[i] = (_Float16)0.f;
    h0[49 * 64 + i] = (_Float16)0.f;
  }
#if defined(__has_builtin) && __has_builtin(__builtin_amdgcn_s_wait_tensorcnt)
  __builtin_amdgcn_s_wait_tensorcnt(0);
#endif
  __syncthreads();

  // stage 1: h0 = x @ W_in + b_in          (12 WMMA)
  for (int mt = 0; mt < 3; ++mt)
    for (int nt = 0; nt < 4; ++nt) {
      v16h a = frag_a_lds_f32(xw, 32, mt * 16, 0, lane);
      v16h b = frag_b_sw(Wp_in, 4, 0, nt, lane);
      v8f c = {0.f, 0.f, 0.f, 0.f, 0.f, 0.f, 0.f, 0.f};
      c = wmma16(a, b, c);
      int n = nt * 16 + n15;
      float bias = b_in[n];
#pragma unroll
      for (int r = 0; r < 8; ++r)
        h0[(1 + mt * 16 + r + 8 * hi) * 64 + n] = (_Float16)(c[r] + bias);
    }
  __syncthreads();

  // stage 2: h1 = relu(conv(h0) + b_conv) as 48x192 @ 192x64   (72 WMMA)
  for (int mt = 0; mt < 3; ++mt)
    for (int nt = 0; nt < 4; ++nt) {
      v8f c = {0.f, 0.f, 0.f, 0.f, 0.f, 0.f, 0.f, 0.f};
#pragma unroll
      for (int kc = 0; kc < 6; ++kc) {
        v16h a = frag_a_conv(h0, mt * 16, kc * 32, lane);
        v16h b = frag_b_sw(Wp_conv, 4, kc, nt, lane);
        c = wmma16(a, b, c);
      }
      int n = nt * 16 + n15;
      float bias = b_conv[n];
#pragma unroll
      for (int r = 0; r < 8; ++r) {
        float z = c[r] + bias;
        h1[(mt * 16 + r + 8 * hi) * 64 + n] = (_Float16)(z > 0.f ? z : 0.f);
      }
    }
  __syncthreads();

  // stage 3: k = h1@Wk + bk, v = h1@Wv + bv   (24+24 WMMA)
  for (int which = 0; which < 2; ++which) {
    const _Float16* W = which ? Wp_v : Wp_k;
    const float* bvec = which ? bv : bk;
    _Float16* dst = which ? vb : kb;
    for (int mt = 0; mt < 3; ++mt)
      for (int nt = 0; nt < 4; ++nt) {
        v8f c = {0.f, 0.f, 0.f, 0.f, 0.f, 0.f, 0.f, 0.f};
#pragma unroll
        for (int kc = 0; kc < 2; ++kc) {
          v16h a = frag_a_lds(h1, 64, mt * 16, kc * 32, lane);
          v16h b = frag_b_sw(W, 4, kc, nt, lane);
          c = wmma16(a, b, c);
        }
        int n = nt * 16 + n15;
        float bias = bvec[n];
#pragma unroll
        for (int r = 0; r < 8; ++r)
          dst[(mt * 16 + r + 8 * hi) * 64 + n] = (_Float16)(c[r] + bias);
      }
  }

  // stage 4: q, last M-tile only (rows 32..47); keep global row 47
  // (= tile row 15 -> VGPR 7 of lanes 16..31).               (8 WMMA)
  for (int nt = 0; nt < 4; ++nt) {
    v8f c = {0.f, 0.f, 0.f, 0.f, 0.f, 0.f, 0.f, 0.f};
#pragma unroll
    for (int kc = 0; kc < 2; ++kc) {
      v16h a = frag_a_lds(h1, 64, 32, kc * 32, lane);
      v16h b = frag_b_sw(Wp_q, 4, kc, nt, lane);
      c = wmma16(a, b, c);
    }
    int n = nt * 16 + n15;
    if (hi == 1) qls[wv][n] = c[7] + bq[n];
  }
  __syncthreads();

  // stage 5: row-47 scores + wave32 softmax (48 columns over 32 lanes)
  float s0, s1 = -1e30f;
  {
    float acc = 0.f;
#pragma unroll
    for (int d = 0; d < 64; ++d) acc += qls[wv][d] * (float)kb[lane * 64 + d];
    s0 = acc * 0.125f;  // 1/sqrt(64)
  }
  if (lane < 16) {
    float acc = 0.f;
#pragma unroll
    for (int d = 0; d < 64; ++d)
      acc += qls[wv][d] * (float)kb[(lane + 32) * 64 + d];
    s1 = acc * 0.125f;
  }
  float mx = fmaxf(s0, s1);
  for (int off = 16; off > 0; off >>= 1) mx = fmaxf(mx, __shfl_xor(mx, off, 32));
  float e0 = __expf(s0 - mx);
  float e1 = (lane < 16) ? __expf(s1 - mx) : 0.f;
  float sm = e0 + e1;
  for (int off = 16; off > 0; off >>= 1) sm += __shfl_xor(sm, off, 32);
  float inv = 1.f / sm;
  ats[wv][lane] = e0 * inv;
  if (lane < 16) ats[wv][lane + 32] = e1 * inv;
  __syncthreads();

  // stage 6: ctx[47] = attn @ v  (each lane: channels d and d+32)
#pragma unroll
  for (int dd = 0; dd < 2; ++dd) {
    int d = lane + dd * 32;
    float acc = 0.f;
#pragma unroll 4
    for (int s = 0; s < 48; ++s) acc += ats[wv][s] * (float)vb[s * 64 + d];
    ctxbuf[seq * 64 + d] = acc;
  }
}

// ---------------------------------------------------------------------------
// Kernel 2: MLP head 64 -> 256 -> 128 -> 12 on all 32768 last-contexts.
// One 16-row M-tile per wave; 4 waves per block (static LDS 56KB).
// ---------------------------------------------------------------------------
#define WAVES_B 4

__global__ __launch_bounds__(32 * WAVES_B)
void mlp_head(const float* __restrict__ ctxbuf,
              const float* __restrict__ b1, const float* __restrict__ b2,
              const float* __restrict__ b3,
              const _Float16* __restrict__ Wp1,
              const _Float16* __restrict__ Wp2,
              const _Float16* __restrict__ Wp3,
              float* __restrict__ out) {
  __shared__ _Float16 xt[WAVES_B][16 * 64];
  __shared__ _Float16 z1[WAVES_B][16 * 256];
  __shared__ _Float16 z2[WAVES_B][16 * 128];

  const int lane = threadIdx.x & 31;
  const int wv   = threadIdx.x >> 5;
  const int hi   = lane >> 4;
  const int n15  = lane & 15;
  const int row0 = (blockIdx.x * WAVES_B + wv) * 16;

  if (threadIdx.x < 16) __builtin_prefetch(Wp2 + (int)threadIdx.x * 4096, 0, 1);

  for (int i = lane; i < 16 * 64; i += 32)
    xt[wv][i] = (_Float16)ctxbuf[(size_t)row0 * 64 + i];
  __syncthreads();

  // z1 = relu(x @ W1 + b1)   (32 WMMA)
  for (int nt = 0; nt < 16; ++nt) {
    v8f c = {0.f, 0.f, 0.f, 0.f, 0.f, 0.f, 0.f, 0.f};
#pragma unroll
    for (int kc = 0; kc < 2; ++kc) {
      v16h a = frag_a_lds(xt[wv], 64, 0, kc * 32, lane);
      v16h b = frag_b_sw(Wp1, 16, kc, nt, lane);
      c = wmma16(a, b, c);
    }
    int n = nt * 16 + n15;
    float bias = b1[n];
#pragma unroll
    for (int r = 0; r < 8; ++r) {
      float z = c[r] + bias;
      z1[wv][(r + 8 * hi) * 256 + n] = (_Float16)(z > 0.f ? z : 0.f);
    }
  }
  __syncthreads();

  // z2 = relu(z1 @ W2 + b2)  (64 WMMA)
  for (int nt = 0; nt < 8; ++nt) {
    v8f c = {0.f, 0.f, 0.f, 0.f, 0.f, 0.f, 0.f, 0.f};
#pragma unroll
    for (int kc = 0; kc < 8; ++kc) {
      v16h a = frag_a_lds(z1[wv], 256, 0, kc * 32, lane);
      v16h b = frag_b_sw(Wp2, 8, kc, nt, lane);
      c = wmma16(a, b, c);
    }
    int n = nt * 16 + n15;
    float bias = b2[n];
#pragma unroll
    for (int r = 0; r < 8; ++r) {
      float z = c[r] + bias;
      z2[wv][(r + 8 * hi) * 128 + n] = (_Float16)(z > 0.f ? z : 0.f);
    }
  }
  __syncthreads();

  // pred = z2 @ W3(pad16) + b3, scatter to out[b][0][p][n]   (4 WMMA)
  {
    v8f c = {0.f, 0.f, 0.f, 0.f, 0.f, 0.f, 0.f, 0.f};
#pragma unroll
    for (int kc = 0; kc < 4; ++kc) {
      v16h a = frag_a_lds(z2[wv], 128, 0, kc * 32, lane);
      v16h b = frag_b_sw(Wp3, 1, kc, 0, lane);
      c = wmma16(a, b, c);
    }
    int p = n15;
    if (p < 12) {
      float bias = b3[p];
#pragma unroll
      for (int r = 0; r < 8; ++r) {
        int g = row0 + r + 8 * hi;        // global row = b*2048 + n
        int bidx = g >> 11, n = g & 2047;
        out[((size_t)bidx * 12 + p) * 2048 + n] = c[r] + bias;
      }
    }
  }
}

// ---------------------------------------------------------------------------
// Launch
// ---------------------------------------------------------------------------
extern "C" void kernel_launch(void* const* d_in, const int* in_sizes, int n_in,
                              void* d_out, int out_size, void* d_ws, size_t ws_size,
                              hipStream_t stream) {
  const float* x      = (const float*)d_in[0];
  const float* W_in   = (const float*)d_in[1];
  const float* b_in   = (const float*)d_in[2];
  const float* W_conv = (const float*)d_in[3];
  const float* b_conv = (const float*)d_in[4];
  const float* Wq     = (const float*)d_in[5];
  const float* bq     = (const float*)d_in[6];
  const float* Wk     = (const float*)d_in[7];
  const float* bk     = (const float*)d_in[8];
  const float* Wv     = (const float*)d_in[9];
  const float* bv     = (const float*)d_in[10];
  const float* W1     = (const float*)d_in[11];
  const float* b1     = (const float*)d_in[12];
  const float* W2     = (const float*)d_in[13];
  const float* b2     = (const float*)d_in[14];
  const float* W3     = (const float*)d_in[15];
  const float* b3     = (const float*)d_in[16];
  float* out = (float*)d_out;

  char* ws = (char*)d_ws;
  _Float16* p_in   = (_Float16*)(ws);            //  32*64  *2 =  4096 B
  _Float16* p_conv = (_Float16*)(ws + 4096);     // 192*64  *2 = 24576 B
  _Float16* p_q    = (_Float16*)(ws + 28672);    //  64*64  *2 =  8192 B
  _Float16* p_k    = (_Float16*)(ws + 36864);
  _Float16* p_v    = (_Float16*)(ws + 45056);
  _Float16* p1     = (_Float16*)(ws + 53248);    //  64*256 *2 = 32768 B
  _Float16* p2     = (_Float16*)(ws + 86016);    // 256*128 *2 = 65536 B
  _Float16* p3     = (_Float16*)(ws + 151552);   // 128*16  *2 =  4096 B
  float*    ctxbuf = (float*)(ws + 155648);      // 32768*64*4 =  8 MB

  prep_weights<<<64, 256, 0, stream>>>(W_in, W_conv, Wq, Wk, Wv, W1, W2, W3,
                                       p_in, p_conv, p_q, p_k, p_v, p1, p2, p3);

  // 16*2048 = 32768 sequences, 2 per block
  attn_fused<<<32768 / WAVES_A, 32 * WAVES_A, 0, stream>>>(
      x, b_in, b_conv, bq, bk, bv, p_in, p_conv, p_q, p_k, p_v, ctxbuf);

  // 32768 rows / 16 per wave / 4 waves per block
  mlp_head<<<32768 / (16 * WAVES_B), 32 * WAVES_B, 0, stream>>>(
      ctxbuf, b1, b2, b3, p1, p2, p3, out);
}
